// BertSelfAttention_59983513256154
// MI455X (gfx1250) — 8-bit matrix operand order repaired and checked by reading, compiled, not run on hardware
//
#include <hip/hip_runtime.h>

typedef __attribute__((ext_vector_type(8))) int v8i;
typedef int v4i_g __attribute__((vector_size(16)));

#if defined(__AMDGCN__) &&                                                   \
    __has_builtin(__builtin_amdgcn_global_load_async_to_lds_b128) &&         \
    __has_builtin(__builtin_amdgcn_s_wait_asynccnt)
#define ASYNC_LDS_OK 1
#define GLOBAL_ASYNC_B128(g, l)                                              \
  __builtin_amdgcn_global_load_async_to_lds_b128(                            \
      (__attribute__((address_space(1))) v4i_g*)(g),                         \
      (__attribute__((address_space(3))) v4i_g*)(l), 0, 0)
#endif

#if defined(__AMDGCN__) && __has_builtin(__builtin_amdgcn_s_wait_dscnt)
#define PB_WAIT_OK 1
#endif


__device__ __forceinline__ float scale_from_bits(unsigned bits) {
  return fmaxf(__uint_as_float(bits) * (1.0f / 127.0f), 1e-8f);
}

__device__ __forceinline__ v8i wmma_iu8(v8i a, v8i b, v8i c) {
  return __builtin_amdgcn_wmma_i32_16x16x64_iu8(true, a, true, b, c, false, false);
}

__device__ __forceinline__ v8i load_a_frag(const signed char* base, int ld) {
  const int lane = threadIdx.x & 31;
  const int m = lane & 15;
  const int half = lane >> 4;
  const signed char* row = base + m * ld + half * 16;
  v8i a;
#pragma unroll
  for (int j = 0; j < 4; ++j) {
    int2 d = *(const int2*)(row + (j >> 1) * 32 + (j & 1) * 8);
    a[2 * j] = d.x;
    a[2 * j + 1] = d.y;
  }
  return a;
}

__device__ __forceinline__ v8i load_b_frag(const signed char* base, int ld) {
  const int lane = threadIdx.x & 31;
  const int n = lane & 15;
  const int half = lane >> 4;
  const signed char* col = base + n * ld + half * 16;
  int4 lo = *(const int4*)(col);
  int4 hi = *(const int4*)(col + 32);
  v8i b;
  b[0] = lo.x; b[1] = lo.y; b[2] = lo.z; b[3] = lo.w;
  b[4] = hi.x; b[5] = hi.y; b[6] = hi.z; b[7] = hi.w;
  return b;
}

__global__ void k_init(unsigned* meta) {
  if (threadIdx.x < 7) meta[threadIdx.x] = 0u;
  if (threadIdx.x == 7) meta[7] = 0x7F800000u;
}

__global__ __launch_bounds__(256) void k_absmax(const float* __restrict__ x,
                                                size_t n, unsigned* out) {
  __shared__ float red[256];
  float lm = 0.0f;
  for (size_t i = (size_t)blockIdx.x * blockDim.x + threadIdx.x; i < n;
       i += (size_t)gridDim.x * blockDim.x)
    lm = fmaxf(lm, fabsf(x[i]));
  red[threadIdx.x] = lm;
  __syncthreads();
  for (int s = 128; s > 0; s >>= 1) {
    if ((int)threadIdx.x < s)
      red[threadIdx.x] = fmaxf(red[threadIdx.x], red[threadIdx.x + s]);
    __syncthreads();
  }
  if (threadIdx.x == 0) atomicMax(out, __float_as_uint(red[0]));
}

__global__ __launch_bounds__(256) void k_quant(const float* __restrict__ x,
                                               size_t n,
                                               const unsigned* __restrict__ amax,
                                               signed char* __restrict__ out) {
  const float inv = 1.0f / scale_from_bits(*amax);
  for (size_t i = (size_t)blockIdx.x * blockDim.x + threadIdx.x; i < n;
       i += (size_t)gridDim.x * blockDim.x) {
    float q = rintf(x[i] * inv);
    q = fminf(fmaxf(q, -128.0f), 127.0f);
    out[i] = (signed char)q;
  }
}

__global__ __launch_bounds__(256) void k_gemm_qkv(
    const signed char* __restrict__ A, const signed char* __restrict__ W,
    const float* __restrict__ bias, unsigned* meta, int slotA, int slotW,
    int slotOut, float* __restrict__ out) {
  __shared__ __align__(16) signed char As[128 * 64];
  __shared__ __align__(16) signed char Bs[64 * 64];
  __shared__ float red[256];

  const int tid = threadIdx.x;
  const int wave = tid >> 5;
  const int lane = tid & 31;
  const int nn = lane & 15;
  const int half = lane >> 4;
  const int wm = wave >> 1;
  const int wn = wave & 1;
  const int m0 = blockIdx.x * 128;
  const int n0 = blockIdx.y * 64;

  const float sab = scale_from_bits(meta[slotA]) * scale_from_bits(meta[slotW]);

  v8i acc[2][2] = {};

  const int r = tid >> 2;
  const int bo = (tid & 3) * 16;
  const signed char* gA0 = A + (size_t)(m0 + r) * 1024 + bo;
  const signed char* gA1 = A + (size_t)(m0 + r + 64) * 1024 + bo;
  const signed char* gB = W + (size_t)(n0 + r) * 1024 + bo;

  for (int k0 = 0; k0 < 1024; k0 += 64) {
#if defined(ASYNC_LDS_OK)
    GLOBAL_ASYNC_B128(gA0 + k0, As + r * 64 + bo);
    GLOBAL_ASYNC_B128(gA1 + k0, As + (r + 64) * 64 + bo);
    GLOBAL_ASYNC_B128(gB + k0, Bs + r * 64 + bo);
#else
    *(int4*)(As + r * 64 + bo) = *(const int4*)(gA0 + k0);
    *(int4*)(As + (r + 64) * 64 + bo) = *(const int4*)(gA1 + k0);
    *(int4*)(Bs + r * 64 + bo) = *(const int4*)(gB + k0);
#endif
    if (k0 + 64 < 1024) {
      __builtin_prefetch(gA0 + k0 + 64, 0, 1);
      __builtin_prefetch(gA1 + k0 + 64, 0, 1);
      __builtin_prefetch(gB + k0 + 64, 0, 1);
    }
#if defined(ASYNC_LDS_OK)
    __builtin_amdgcn_s_wait_asynccnt(0);
#endif
    __syncthreads();

    v8i af0 = load_a_frag(As + (wm * 32) * 64, 64);
    v8i af1 = load_a_frag(As + (wm * 32 + 16) * 64, 64);
    v8i bf0 = load_b_frag(Bs + (wn * 32) * 64, 64);
    v8i bf1 = load_b_frag(Bs + (wn * 32 + 16) * 64, 64);
    acc[0][0] = wmma_iu8(af0, bf0, acc[0][0]);
    acc[0][1] = wmma_iu8(af0, bf1, acc[0][1]);
    acc[1][0] = wmma_iu8(af1, bf0, acc[1][0]);
    acc[1][1] = wmma_iu8(af1, bf1, acc[1][1]);
    __syncthreads();
  }

  float lmax = 0.0f;
#pragma unroll
  for (int i = 0; i < 2; ++i) {
#pragma unroll
    for (int j = 0; j < 2; ++j) {
#pragma unroll
      for (int rr = 0; rr < 8; ++rr) {
        const int gm = m0 + wm * 32 + i * 16 + rr + 8 * half;
        const int gn = n0 + wn * 32 + j * 16 + nn;
        const float v = (float)acc[i][j][rr] * sab + bias[gn];
        const int b = gm >> 9, s = gm & 511;
        const int h = gn >> 6, d = gn & 63;
        out[(((size_t)(b * 16 + h) * 512) + s) * 64 + d] = v;
        lmax = fmaxf(lmax, fabsf(v));
      }
    }
  }
  red[tid] = lmax;
  __syncthreads();
  for (int s2 = 128; s2 > 0; s2 >>= 1) {
    if (tid < s2) red[tid] = fmaxf(red[tid], red[tid + s2]);
    __syncthreads();
  }
  if (tid == 0) atomicMax(&meta[slotOut], __float_as_uint(red[0]));
}

__global__ __launch_bounds__(256) void k_stats(
    const signed char* __restrict__ qi, const signed char* __restrict__ ki,
    const float* __restrict__ mask, unsigned* meta, float* __restrict__ mrow,
    float* __restrict__ lrow) {
  const int tid = threadIdx.x, wave = tid >> 5, lane = tid & 31;
  const int nn = lane & 15, half = lane >> 4;
  const int bh = blockIdx.x >> 2;
  const int q0 = (blockIdx.x & 3) * 128 + wave * 16;
  const int b = bh >> 4;
  const float sqk =
      scale_from_bits(meta[4]) * scale_from_bits(meta[5]) * 0.125f;
  const signed char* qp = qi + (size_t)bh * 512 * 64;
  const signed char* kp = ki + (size_t)bh * 512 * 64;
  const float* mb = mask + (size_t)b * 512;

  v8i aq = load_a_frag(qp + q0 * 64, 64);
  float m[8], l[8];
#pragma unroll
  for (int r = 0; r < 8; ++r) { m[r] = -3.0e38f; l[r] = 0.0f; }

  for (int kk = 0; kk < 512; kk += 16) {
    v8i bk = load_b_frag(kp + kk * 64, 64);
    v8i sc = {};
    sc = wmma_iu8(aq, bk, sc);
    const float mkv = mb[kk + nn];
#pragma unroll
    for (int r = 0; r < 8; ++r) {
      const float s = (float)sc[r] * sqk + mkv;
      float t = s;
      t = fmaxf(t, __shfl_xor(t, 1));
      t = fmaxf(t, __shfl_xor(t, 2));
      t = fmaxf(t, __shfl_xor(t, 4));
      t = fmaxf(t, __shfl_xor(t, 8));
      const float mn = fmaxf(m[r], t);
      float p = __expf(s - mn);
      p += __shfl_xor(p, 1);
      p += __shfl_xor(p, 2);
      p += __shfl_xor(p, 4);
      p += __shfl_xor(p, 8);
      l[r] = l[r] * __expf(m[r] - mn) + p;
      m[r] = mn;
    }
  }
  if (nn == 0) {
#pragma unroll
    for (int r = 0; r < 8; ++r) {
      const size_t row = (size_t)bh * 512 + q0 + r + 8 * half;
      mrow[row] = m[r];
      lrow[row] = l[r];
      atomicMin(&meta[7], __float_as_uint(l[r]));
    }
  }
}

__global__ __launch_bounds__(256) void k_attn_av(
    const signed char* __restrict__ qi, const signed char* __restrict__ ki,
    const signed char* __restrict__ vi, const float* __restrict__ mask,
    const unsigned* __restrict__ meta, const float* __restrict__ mrow,
    const float* __restrict__ lrow, float* __restrict__ out) {
  __shared__ __align__(16) signed char vt[64 * 512];
  __shared__ __align__(16) signed char pb[8 * 16 * 64];

  const int tid = threadIdx.x, wave = tid >> 5, lane = tid & 31;
  const int nn = lane & 15, half = lane >> 4;
  const int bh = blockIdx.x >> 2;
  const int q0 = (blockIdx.x & 3) * 128 + wave * 16;
  const int b = bh >> 4, h = bh & 15;

  const float sqk = scale_from_bits(meta[4]) * scale_from_bits(meta[5]) * 0.125f;
  const float pmax = 1.0f / __uint_as_float(meta[7]);
  const float sp = fmaxf(pmax * (1.0f / 127.0f), 1e-8f);
  const float inv_sp = 1.0f / sp;
  const float spv = sp * scale_from_bits(meta[6]);

  const signed char* qp = qi + (size_t)bh * 512 * 64;
  const signed char* kp = ki + (size_t)bh * 512 * 64;
  const signed char* vp = vi + (size_t)bh * 512 * 64;
  const float* mb = mask + (size_t)b * 512;

  {
    const int* v32 = (const int*)vp;
#pragma unroll
    for (int i = 0; i < 32; ++i) {
      const int idx = tid + i * 256;
      const int key = idx >> 4;
      const int d4 = (idx & 15) * 4;
      const int w = v32[idx];
      vt[(d4 + 0) * 512 + key] = (signed char)(w & 0xff);
      vt[(d4 + 1) * 512 + key] = (signed char)((w >> 8) & 0xff);
      vt[(d4 + 2) * 512 + key] = (signed char)((w >> 16) & 0xff);
      vt[(d4 + 3) * 512 + key] = (signed char)((w >> 24) & 0xff);
    }
  }
  __syncthreads();

  v8i aq = load_a_frag(qp + q0 * 64, 64);
  float mr[8], il[8];
#pragma unroll
  for (int r = 0; r < 8; ++r) {
    const size_t row = (size_t)bh * 512 + q0 + r + 8 * half;
    mr[r] = mrow[row];
    il[r] = 1.0f / lrow[row];
  }

  v8i acc[4] = {};
  signed char* pw = pb + wave * (16 * 64);

  for (int kk = 0; kk < 512; kk += 64) {
#pragma unroll
    for (int t = 0; t < 4; ++t) {
      v8i bk = load_b_frag(kp + (kk + t * 16) * 64, 64);
      v8i sc = {};
      sc = wmma_iu8(aq, bk, sc);
      const float mkv = mb[kk + t * 16 + nn];
#pragma unroll
      for (int r = 0; r < 8; ++r) {
        const float s = (float)sc[r] * sqk + mkv;
        const float p = __expf(s - mr[r]) * il[r];
        int q8 = (int)rintf(p * inv_sp);
        q8 = q8 < 0 ? 0 : (q8 > 127 ? 127 : q8);
        pw[(r + 8 * half) * 64 + t * 16 + nn] = (signed char)q8;
      }
    }
#if defined(PB_WAIT_OK)
    __builtin_amdgcn_s_wait_dscnt(0);
#else
    __syncthreads();
#endif
    v8i ap = load_a_frag(pw, 64);
#pragma unroll
    for (int g = 0; g < 4; ++g) {
      v8i bv = load_b_frag(vt + (g * 16) * 512 + kk, 512);
      acc[g] = wmma_iu8(ap, bv, acc[g]);
    }
#if !defined(PB_WAIT_OK)
    __syncthreads();
#endif
  }

#pragma unroll
  for (int g = 0; g < 4; ++g) {
#pragma unroll
    for (int r = 0; r < 8; ++r) {
      const int s = q0 + r + 8 * half;
      const int d = g * 16 + nn;
      out[((size_t)b * 512 + s) * 1024 + h * 64 + d] = (float)acc[g][r] * spv;
    }
  }
}

extern "C" void kernel_launch(void* const* d_in, const int* in_sizes, int n_in,
                              void* d_out, int out_size, void* d_ws,
                              size_t ws_size, hipStream_t stream) {
  const float* hs = (const float*)d_in[0];
  const float* mask = (const float*)d_in[1];
  const float* Wq = (const float*)d_in[2];
  const float* bq = (const float*)d_in[3];
  const float* Wk = (const float*)d_in[4];
  const float* bk = (const float*)d_in[5];
  const float* Wv = (const float*)d_in[6];
  const float* bv = (const float*)d_in[7];
  float* out = (float*)d_out;

  char* ws = (char*)d_ws;
  const size_t nHS = (size_t)16 * 512 * 1024;
  const size_t nW = (size_t)1024 * 1024;

  unsigned* meta = (unsigned*)ws;
  size_t off = 256;
  signed char* hs8 = (signed char*)(ws + off); off += nHS;
  signed char* wq8 = (signed char*)(ws + off); off += nW;
  signed char* wk8 = (signed char*)(ws + off); off += nW;
  signed char* wv8 = (signed char*)(ws + off); off += nW;
  float* qf = (float*)(ws + off); off += nHS * 4;
  float* kf = (float*)(ws + off); off += nHS * 4;
  float* vf = (float*)(ws + off); off += nHS * 4;
  signed char* q8 = (signed char*)(ws + off); off += nHS;
  signed char* k8 = (signed char*)(ws + off); off += nHS;
  signed char* v8 = (signed char*)(ws + off); off += nHS;
  float* mrow = (float*)(ws + off); off += (size_t)256 * 512 * 4;
  float* lrow = (float*)(ws + off); off += (size_t)256 * 512 * 4;

  k_init<<<1, 32, 0, stream>>>(meta);
  k_absmax<<<1024, 256, 0, stream>>>(hs, nHS, meta + 0);
  k_absmax<<<256, 256, 0, stream>>>(Wq, nW, meta + 1);
  k_absmax<<<256, 256, 0, stream>>>(Wk, nW, meta + 2);
  k_absmax<<<256, 256, 0, stream>>>(Wv, nW, meta + 3);
  k_quant<<<2048, 256, 0, stream>>>(hs, nHS, meta + 0, hs8);
  k_quant<<<512, 256, 0, stream>>>(Wq, nW, meta + 1, wq8);
  k_quant<<<512, 256, 0, stream>>>(Wk, nW, meta + 2, wk8);
  k_quant<<<512, 256, 0, stream>>>(Wv, nW, meta + 3, wv8);

  dim3 gg(64, 16);
  k_gemm_qkv<<<gg, 256, 0, stream>>>(hs8, wq8, bq, meta, 0, 1, 4, qf);
  k_gemm_qkv<<<gg, 256, 0, stream>>>(hs8, wk8, bk, meta, 0, 2, 5, kf);
  k_gemm_qkv<<<gg, 256, 0, stream>>>(hs8, wv8, bv, meta, 0, 3, 6, vf);

  k_quant<<<2048, 256, 0, stream>>>(qf, nHS, meta + 4, q8);
  k_quant<<<2048, 256, 0, stream>>>(kf, nHS, meta + 5, k8);
  k_quant<<<2048, 256, 0, stream>>>(vf, nHS, meta + 6, v8);

  k_stats<<<1024, 256, 0, stream>>>(q8, k8, mask, meta, mrow, lrow);
  k_attn_av<<<1024, 256, 0, stream>>>(q8, k8, v8, mask, meta, mrow, lrow, out);

  (void)in_sizes; (void)n_in; (void)out_size; (void)ws_size;
}
